// SelfAttention_64476049048220
// MI455X (gfx1250) — compile-verified, matrix operand order checked by reading, not run on hardware
//
#include <hip/hip_runtime.h>
#include <hip/hip_bf16.h>

// ---------------------------------------------------------------------------
// Self-attention (softmax over QUERY axis) + residual + LayerNorm, CDNA5.
// GEMMs: v_wmma_f32_16x16x32_f16, operands staged in LDS by the Tensor Data
// Mover (tensor_load_to_lds, double-buffered, TDM LDS-padding for bank-
// conflict-free ds_load_b128 fragment reads).
// Workspace requirement: ~220 MB (see kernel_launch layout).
// ---------------------------------------------------------------------------

typedef _Float16 h16;
typedef __attribute__((ext_vector_type(16))) _Float16       v16h;
typedef __attribute__((ext_vector_type(8)))  float          v8f;
typedef __attribute__((ext_vector_type(8)))  unsigned short u16x8;
typedef __attribute__((ext_vector_type(4)))  unsigned int   u32x4;
typedef __attribute__((ext_vector_type(8)))  unsigned int   u32x8;

#define D_DIM 768
#define T_DIM 2048
#define B_DIM 16

// LDS tile geometry: rows are 32 f16 (64B) + 16B TDM pad -> 40 f16 (80B) pitch
#define TROW 40
#define A_ROWS 128
#define B_ROWS 64
#define SA_ELEMS (2 * A_ROWS * TROW)   // double-buffered A tile
#define SB_ELEMS (2 * B_ROWS * TROW)   // double-buffered B tile

// ---- TDM: DMA one [rows x 32] f16 tile (row stride = strideElems) to LDS ---
// D# group0: count=1 | lds_addr | global_addr(57b) | type=2.
// D# group1: data_size=2B, pad_enable, pad_interval=16 DW, pad_amount=4 DW,
//            tensor_dim0=strideElems, tensor_dim1=rows, tile_dim0=32,
//            tile_dim1=rows, tensor_dim0_stride=strideElems.
__device__ __forceinline__ void tdm_load_tile(unsigned lds_off,
                                              const h16* __restrict__ gptr,
                                              unsigned rows,
                                              unsigned strideElems) {
  const unsigned long long ga = (unsigned long long)(const void*)gptr;
  u32x4 g0;
  g0[0] = 1u;                                        // count=1, user mode
  g0[1] = lds_off;                                   // lds_addr  [63:32]
  g0[2] = (unsigned)(ga & 0xffffffffu);              // gaddr lo  [95:64]
  g0[3] = (unsigned)((ga >> 32) & 0x01ffffffu)       // gaddr hi  [120:96]
        | (2u << 30);                                // type=2 (image)
  u32x8 g1;
  g1[0] = (1u << 16)                                 // data_size = 2 bytes
        | (1u << 20)                                 // pad_enable
        | (3u << 22)                                 // pad_interval = 16 DW (64B)
        | (3u << 25);                                // pad_amount   =  4 DW (16B)
  g1[1] = (strideElems & 0xffffu) << 16;             // tensor_dim0 [63:48]
  g1[2] = (strideElems >> 16)                        // tensor_dim0 [79:64]
        | ((rows & 0xffffu) << 16);                  // tensor_dim1 [95:80]
  g1[3] = (rows >> 16)                               // tensor_dim1 [111:96]
        | (32u << 16);                               // tile_dim0 = 32 elems
  g1[4] = rows & 0xffffu;                            // tile_dim1; tile_dim2=0
  g1[5] = strideElems;                               // tensor_dim0_stride lo32
  g1[6] = 0u;                                        // stride hi / dim1_stride lo
  g1[7] = 0u;
  asm volatile("tensor_load_to_lds %0, %1" :: "s"(g0), "s"(g1) : "memory");
}

// ---- 32B fragment gather from an LDS tile row (two ds_load_b128) -----------
__device__ __forceinline__ v16h lds_frag(const h16* __restrict__ p) {
  v16h f;
  __builtin_memcpy(&f, p, 16);                                   // K=+0..7
  __builtin_memcpy(reinterpret_cast<char*>(&f) + 16, p + 16, 16); // K=+16..23
  return f;
}

// ---- TDM-staged, double-buffered WMMA core ---------------------------------
// Block = 256 thr = 8 waves (4 M x 2 N), block tile 128x64, wave tile 32x32.
// Ab points at the block's 128 A rows; Bb at the block's 64 B rows ([N,K]).
__device__ __forceinline__ void gemm_core_tdm(const h16* __restrict__ Ab,
                                              const h16* __restrict__ Bb,
                                              int K, h16* sm,
                                              int lane, int wid, v8f acc[4]) {
  h16* sA = sm;
  h16* sB = sm + SA_ELEMS;
  const unsigned ldsA0 = (unsigned)(unsigned long long)(void*)sA;
  const unsigned ldsB0 = (unsigned)(unsigned long long)(void*)sB;
  const int m0in = (wid >> 1) * 32;     // wave M offset inside block tile
  const int n0in = (wid & 1) * 32;      // wave N offset inside block tile
  const int KT = K / 32;

  if (wid == 0) {                       // prologue: stage k-tile 0 into buf 0
    tdm_load_tile(ldsA0, Ab, A_ROWS, (unsigned)K);
    tdm_load_tile(ldsB0, Bb, B_ROWS, (unsigned)K);
  }

  for (int kt = 0; kt < KT; ++kt) {
    const int buf = kt & 1;
    if (wid == 0) {
      if (kt + 1 < KT) {                // stage next k-tile, then wait for cur
        const int nb = buf ^ 1;
        tdm_load_tile(ldsA0 + (unsigned)(nb * A_ROWS * TROW * 2),
                      Ab + (long)(kt + 1) * 32, A_ROWS, (unsigned)K);
        tdm_load_tile(ldsB0 + (unsigned)(nb * B_ROWS * TROW * 2),
                      Bb + (long)(kt + 1) * 32, B_ROWS, (unsigned)K);
        __builtin_amdgcn_s_wait_tensorcnt(2);   // cur 2 DMAs done, next in flight
      } else {
        __builtin_amdgcn_s_wait_tensorcnt(0);
      }
    }
    __syncthreads();                    // publish LDS tile to all waves

    // A-matrix 16x32 layout (ISA 7.12.2): lanes 0-15 row=lane, K 0..7/16..23;
    // lanes 16-31 same rows, K 8..15/24..31. B uses same map over N.
    const h16* pa0 = sA + buf * A_ROWS * TROW
                   + (m0in + (lane & 15)) * TROW + ((lane >> 4) << 3);
    const h16* pb0 = sB + buf * B_ROWS * TROW
                   + (n0in + (lane & 15)) * TROW + ((lane >> 4) << 3);
    v16h a0 = lds_frag(pa0);
    v16h a1 = lds_frag(pa0 + 16 * TROW);
    v16h b0 = lds_frag(pb0);
    v16h b1 = lds_frag(pb0 + 16 * TROW);
    acc[0] = __builtin_amdgcn_wmma_f32_16x16x32_f16(false, a0, false, b0, (short)0, acc[0], false, false);
    acc[1] = __builtin_amdgcn_wmma_f32_16x16x32_f16(false, a0, false, b1, (short)0, acc[1], false, false);
    acc[2] = __builtin_amdgcn_wmma_f32_16x16x32_f16(false, a1, false, b0, (short)0, acc[2], false, false);
    acc[3] = __builtin_amdgcn_wmma_f32_16x16x32_f16(false, a1, false, b1, (short)0, acc[3], false, false);

    __syncthreads();                    // all reads done before buf is re-DMAed
  }
}

// ---- GEMM: C(f16) = A[M,K] * B[N,K]^T + bias, optional transposed store ----
// grid: (N/64, M/128, batches); block: 256 threads.
__global__ void __launch_bounds__(256)
sa_gemm_f16out(const h16* __restrict__ A, const h16* __restrict__ Bm,
               const float* __restrict__ bias, h16* __restrict__ C,
               int M, int N, int K, long strideA, long strideC,
               int transposeOut) {
  __shared__ h16 smem[SA_ELEMS + SB_ELEMS];
  const int lane = threadIdx.x & 31;
  const int wid  = threadIdx.x >> 5;
  const int m0 = blockIdx.y * 128 + (wid >> 1) * 32;
  const int n0 = blockIdx.x * 64  + (wid & 1) * 32;
  const h16* Ab = A + (long)blockIdx.z * strideA + (long)(blockIdx.y * 128) * K;
  const h16* Bb = Bm + (long)(blockIdx.x * 64) * K;
  h16*       Cb = C + (long)blockIdx.z * strideC;

  v8f acc[4] = {};
  gemm_core_tdm(Ab, Bb, K, smem, lane, wid, acc);

  // C/D layout: VGPR r, lanes 0-15 -> row r, lanes 16-31 -> row r+8; col=lane&15
  auto store_tile = [&](v8f c, int mt, int nt) {
    const int col   = nt + (lane & 15);
    const int rbase = mt + ((lane >> 4) << 3);
    const float bv  = bias ? bias[col] : 0.0f;
    if (!transposeOut) {              // C is [M,N]
#pragma unroll
      for (int r = 0; r < 8; ++r)
        Cb[(long)(rbase + r) * N + col] = (h16)(c[r] + bv);
    } else {                          // C is [N,M]; 8 contiguous: one 16B store
      u16x8 pk;
#pragma unroll
      for (int r = 0; r < 8; ++r) {
        h16 h = (h16)(c[r] + bv);
        pk[r] = __builtin_bit_cast(unsigned short, h);
      }
      __builtin_memcpy(Cb + (long)col * M + rbase, &pk, 16);
    }
  };
  store_tile(acc[0], m0,      n0);
  store_tile(acc[1], m0,      n0 + 16);
  store_tile(acc[2], m0 + 16, n0);
  store_tile(acc[3], m0 + 16, n0 + 16);
}

// ---- GEMM: C(f32) = A[M,K] * B[N,K]^T (+ optional residual) ----------------
__global__ void __launch_bounds__(256)
sa_gemm_f32out(const h16* __restrict__ A, const h16* __restrict__ Bm,
               const float* __restrict__ residual, float* __restrict__ C,
               int M, int N, int K) {
  __shared__ h16 smem[SA_ELEMS + SB_ELEMS];
  const int lane = threadIdx.x & 31;
  const int wid  = threadIdx.x >> 5;
  const int m0 = blockIdx.y * 128 + (wid >> 1) * 32;
  const int n0 = blockIdx.x * 64  + (wid & 1) * 32;
  const h16* Ab = A + (long)(blockIdx.y * 128) * K;
  const h16* Bb = Bm + (long)(blockIdx.x * 64) * K;

  v8f acc[4] = {};
  gemm_core_tdm(Ab, Bb, K, smem, lane, wid, acc);

  auto store_tile = [&](v8f c, int mt, int nt) {
    const int col   = nt + (lane & 15);
    const int rbase = mt + ((lane >> 4) << 3);
#pragma unroll
    for (int r = 0; r < 8; ++r) {
      const long idx = (long)(rbase + r) * N + col;
      float v = c[r];
      if (residual) v += residual[idx];
      C[idx] = v;
    }
  };
  store_tile(acc[0], m0,      n0);
  store_tile(acc[1], m0,      n0 + 16);
  store_tile(acc[2], m0 + 16, n0);
  store_tile(acc[3], m0 + 16, n0 + 16);
}

// ---- f32 -> f16 convert ----------------------------------------------------
__global__ void __launch_bounds__(256)
sa_cvt_f16(const float* __restrict__ src, h16* __restrict__ dst, long n) {
  long i = (long)blockIdx.x * blockDim.x + threadIdx.x;
  const long stride = (long)gridDim.x * blockDim.x;
  for (; i < n; i += stride) dst[i] = (h16)src[i];
}

// ---- column softmax over q (axis=1 of logits[q,t]), * 1/sqrt(D) -----------
// block (64,4): 64 columns, 4 row-strided workers per column; coalesced reads.
__global__ void __launch_bounds__(256)
sa_col_softmax(const float* __restrict__ S, h16* __restrict__ P, float scale) {
  __shared__ float red[4][64];
  const int tx = threadIdx.x, ty = threadIdx.y;
  const int t  = blockIdx.x * 64 + tx;

  float m = -3.4e38f;
  for (int q = ty; q < T_DIM; q += 4) m = fmaxf(m, S[(long)q * T_DIM + t]);
  red[ty][tx] = m;
  __syncthreads();
  m = fmaxf(fmaxf(red[0][tx], red[1][tx]), fmaxf(red[2][tx], red[3][tx]));
  __syncthreads();

  float s = 0.0f;
  for (int q = ty; q < T_DIM; q += 4) s += __expf(S[(long)q * T_DIM + t] - m);
  red[ty][tx] = s;
  __syncthreads();
  s = red[0][tx] + red[1][tx] + red[2][tx] + red[3][tx];

  const float inv = scale / s;
  for (int q = ty; q < T_DIM; q += 4)
    P[(long)q * T_DIM + t] = (h16)(__expf(S[(long)q * T_DIM + t] - m) * inv);
}

// ---- in-place LayerNorm over last dim (768 = 256 threads x 3) --------------
__global__ void __launch_bounds__(256)
sa_layernorm(float* __restrict__ Y, const float* __restrict__ gamma,
             const float* __restrict__ beta) {
  __shared__ float red[8];
  __shared__ float sMean, sVar;
  float* y = Y + (long)blockIdx.x * D_DIM;
  const int tid = threadIdx.x;

  float v0 = y[tid], v1 = y[tid + 256], v2 = y[tid + 512];
  float s = v0 + v1 + v2;
#pragma unroll
  for (int off = 16; off; off >>= 1) s += __shfl_down(s, off, 32);
  if ((tid & 31) == 0) red[tid >> 5] = s;
  __syncthreads();
  if (tid == 0) {
    float tot = 0.0f;
    for (int i = 0; i < 8; ++i) tot += red[i];
    sMean = tot * (1.0f / D_DIM);
  }
  __syncthreads();
  const float mu = sMean;
  const float d0 = v0 - mu, d1 = v1 - mu, d2 = v2 - mu;
  float vs = d0 * d0 + d1 * d1 + d2 * d2;
#pragma unroll
  for (int off = 16; off; off >>= 1) vs += __shfl_down(vs, off, 32);
  if ((tid & 31) == 0) red[tid >> 5] = vs;
  __syncthreads();
  if (tid == 0) {
    float tot = 0.0f;
    for (int i = 0; i < 8; ++i) tot += red[i];
    sVar = tot * (1.0f / D_DIM);
  }
  __syncthreads();
  const float rinv = rsqrtf(sVar + 1e-5f);
  y[tid]       = d0 * rinv * gamma[tid]       + beta[tid];
  y[tid + 256] = d1 * rinv * gamma[tid + 256] + beta[tid + 256];
  y[tid + 512] = d2 * rinv * gamma[tid + 512] + beta[tid + 512];
}

// ---------------------------------------------------------------------------
extern "C" void kernel_launch(void* const* d_in, const int* in_sizes, int n_in,
                              void* d_out, int out_size, void* d_ws, size_t ws_size,
                              hipStream_t stream) {
  (void)in_sizes; (void)n_in; (void)out_size; (void)ws_size;
  const float* x     = (const float*)d_in[0];
  const float* Wk    = (const float*)d_in[1];
  const float* bk    = (const float*)d_in[2];
  const float* Wq    = (const float*)d_in[3];
  const float* bq    = (const float*)d_in[4];
  const float* Wv    = (const float*)d_in[5];
  const float* bv    = (const float*)d_in[6];
  const float* gamma = (const float*)d_in[7];
  const float* beta  = (const float*)d_in[8];
  float* out = (float*)d_out;

  const long BTD = (long)B_DIM * T_DIM * D_DIM;   // 25,165,824
  const long DD  = (long)D_DIM * D_DIM;
  const long TT  = (long)T_DIM * T_DIM;

  // Workspace layout (256B-aligned slabs), ~220 MB total:
  char* ws = (char*)d_ws;
  size_t off = 0;
  auto alloc = [&](size_t bytes) -> void* {
    off = (off + 255) & ~(size_t)255;
    void* p = ws + off;
    off += bytes;
    return p;
  };
  h16*   xh  = (h16*)  alloc(BTD * 2);   // x  as f16          (48 MB)
  h16*   qh  = (h16*)  alloc(BTD * 2);   // Q  as f16          (48 MB)
  h16*   kh  = (h16*)  alloc(BTD * 2);   // K  as f16          (48 MB)
  h16*   vt  = (h16*)  alloc(BTD * 2);   // V^T per batch [D,T](48 MB)
  h16*   wkh = (h16*)  alloc(DD * 2);
  h16*   wqh = (h16*)  alloc(DD * 2);
  h16*   wvh = (h16*)  alloc(DD * 2);
  float* S   = (float*)alloc(TT * 4);    // logits, per-batch reuse (16 MB)
  h16*   P   = (h16*)  alloc(TT * 2);    // probs,  per-batch reuse ( 8 MB)

  // 1) convert inputs to f16
  sa_cvt_f16<<<2048, 256, 0, stream>>>(x,  xh,  BTD);
  sa_cvt_f16<<<256,  256, 0, stream>>>(Wk, wkh, DD);
  sa_cvt_f16<<<256,  256, 0, stream>>>(Wq, wqh, DD);
  sa_cvt_f16<<<256,  256, 0, stream>>>(Wv, wvh, DD);

  // 2) Q, K, V^T = x * W^T + b  (WMMA + TDM, all batches)
  dim3 blk(256);
  dim3 gQKV(D_DIM / 64, T_DIM / 128, B_DIM);   // (12, 16, 16)
  sa_gemm_f16out<<<gQKV, blk, 0, stream>>>(xh, wqh, bq, qh,
      T_DIM, D_DIM, D_DIM, (long)T_DIM * D_DIM, (long)T_DIM * D_DIM, 0);
  sa_gemm_f16out<<<gQKV, blk, 0, stream>>>(xh, wkh, bk, kh,
      T_DIM, D_DIM, D_DIM, (long)T_DIM * D_DIM, (long)T_DIM * D_DIM, 0);
  sa_gemm_f16out<<<gQKV, blk, 0, stream>>>(xh, wvh, bv, vt,
      T_DIM, D_DIM, D_DIM, (long)T_DIM * D_DIM, (long)D_DIM * T_DIM, 1);

  // 3) per batch: logits -> column softmax -> P*V + residual (stream-serial,
  //    so S/P slabs are safely reused)
  const float inv_sqrt_d = 0.03608439182435161f;  // 1/sqrt(768)
  for (int b = 0; b < B_DIM; ++b) {
    const h16* qb  = qh + (long)b * T_DIM * D_DIM;
    const h16* kb  = kh + (long)b * T_DIM * D_DIM;
    const h16* vtb = vt + (long)b * D_DIM * T_DIM;

    dim3 gL(T_DIM / 64, T_DIM / 128, 1);         // (32, 16)
    sa_gemm_f32out<<<gL, blk, 0, stream>>>(qb, kb, nullptr, S,
                                           T_DIM, T_DIM, D_DIM);

    sa_col_softmax<<<dim3(T_DIM / 64, 1, 1), dim3(64, 4, 1), 0, stream>>>(
        S, P, inv_sqrt_d);

    dim3 gPV(D_DIM / 64, T_DIM / 128, 1);        // (12, 16)
    sa_gemm_f32out<<<gPV, blk, 0, stream>>>(P, vtb,
        x + (long)b * T_DIM * D_DIM, out + (long)b * T_DIM * D_DIM,
        T_DIM, D_DIM, T_DIM);
  }

  // 4) LayerNorm in place on d_out
  sa_layernorm<<<B_DIM * T_DIM, 256, 0, stream>>>(out, gamma, beta);
}
